// FeatIterpNFMLP_22428319220266
// MI455X (gfx1250) — matrix operand order repaired and checked by reading, not compiled, not run on hardware
//
#include <hip/hip_runtime.h>

typedef _Float16 half_t;
typedef _Float16 v4h  __attribute__((ext_vector_type(4)));
typedef _Float16 v8h  __attribute__((ext_vector_type(8)));
typedef _Float16 v16h __attribute__((ext_vector_type(16)));
typedef float    v8f  __attribute__((ext_vector_type(8)));

#define RESO   16
#define FEAT   8
#define HID0   64
#define OUTD   16
#define NPTS   8192
#define NEXMP  64
#define TTL    (RESO * RESO * RESO * FEAT)     // 32768 floats per grid

#define WAVES  4
#define TPB    (WAVES * 32)
#define TILES  8                               // 16-point tiles per wave
#define PPB    (WAVES * TILES * 16)            // 512 points per block
#define BPE    (NPTS / PPB)                    // 16 blocks per example

// B-fragment store: fragment = 16 lanes x 16 contiguous halves, lane row
// stride 24 halves (48 B -> conflict-free bank mapping, 16B aligned).
#define FSTR   24
#define FROW   (16 * FSTR)                     // 384 halves per fragment
#define NFRAG  28                              // 8 (L0) + 16 (L1) + 4 (L2)

#define SFSTR  40                              // feature row stride (halves)
#define HSTR   72                              // hidden row stride (halves)

// concat two v8h into a v16h WMMA operand
#define CAT16(LO, HI) __builtin_shufflevector((LO), (HI), \
        0,1,2,3,4,5,6,7,8,9,10,11,12,13,14,15)

__global__ __launch_bounds__(TPB) void featmlp_kernel(
    const int*   __restrict__ idx,
    const float* __restrict__ x,
    const float* __restrict__ emb,
    const float* __restrict__ W0, const float* __restrict__ b0,
    const float* __restrict__ W1, const float* __restrict__ b1,
    const float* __restrict__ Wo, const float* __restrict__ bo,
    float*       __restrict__ out)
{
    __shared__ __align__(16) half_t sBf[NFRAG * FROW];       // weight fragments
    __shared__ __align__(16) half_t sF [WAVES][16][SFSTR];   // features (padded)
    __shared__ __align__(16) half_t sH0[WAVES][16][HSTR];
    __shared__ __align__(16) half_t sH1[WAVES][16][HSTR];
    __shared__ float sB0[HID0], sB1[HID0], sBo[OUTD];

    const int tid = threadIdx.x;

    // ---------------- one-time staging ----------------
    // weights -> f16, pre-swizzled into WMMA B-fragment layout.
    // this code's B element i is (K = hi*16+i, col = ln): contiguous k, same order as A.
    for (int e = tid; e < NFRAG * 256; e += TPB) {
        const int frag = e >> 8, ln = (e >> 4) & 15, i = e & 15;
        float v;
        if (frag < 8) {                         // layer0: frag = fhi*4+n, K pad
            const int fhi = frag >> 2, n = frag & 3, K = fhi * 16 + i;
            v = (K < FEAT) ? W0[K * HID0 + n * 16 + ln] : 0.0f;
        } else if (frag < 24) {                 // layer1: frag = 8+kt*8+fhi*4+n
            const int f = frag - 8, kt = f >> 3, fhi = (f >> 2) & 1, n = f & 3;
            v = W1[(kt * 32 + fhi * 16 + i) * HID0 + n * 16 + ln];
        } else {                                // layer2: frag = 24+kt*2+fhi
            const int f = frag - 24, kt = f >> 1, fhi = f & 1;
            v = Wo[(kt * 32 + fhi * 16 + i) * OUTD + ln];
        }
        sBf[frag * FROW + ln * FSTR + i] = (half_t)v;
    }
    for (int i = tid; i < HID0; i += TPB) { sB0[i] = b0[i]; sB1[i] = b1[i]; }
    for (int i = tid; i < OUTD; i += TPB) sBo[i] = bo[i];
    // zero the feature staging once: K = 8..31 pad is never rewritten
    {
        v8h z = {};
        v8h* p = (v8h*)&sF[0][0][0];
        for (int e = tid; e < WAVES * 16 * (SFSTR / 8); e += TPB) p[e] = z;
    }
    __syncthreads();

    const int wave = tid >> 5;
    const int lane = tid & 31;
    const int ln   = lane & 15;     // A row (point) / B,C,D column
    const int hi   = lane >> 4;     // lane-half selector

    const int bex  = blockIdx.x / BPE;
    const int base = (blockIdx.x % BPE) * PPB;
    const float* g = emb + (size_t)idx[bex] * TTL;

    // per-lane fragment base pointers (lane-uniform layouts, no conditionals)
    const half_t* f0b = sBf + (hi * 4) * FROW + ln * FSTR;           // + n*FROW
    const half_t* f1b = sBf + (8 + hi * 4) * FROW + ln * FSTR;       // + (kt*8+n)*FROW
    const half_t* f2b = sBf + (24 + hi) * FROW + ln * FSTR;          // + kt*2*FROW

#pragma unroll 1
    for (int t = 0; t < TILES; ++t) {
        const int p0 = base + (t * WAVES + wave) * 16;

        // ---------- trilinear interpolation (fp32 math) ----------
        {
            const float* xp = x + ((size_t)bex * NPTS + p0 + ln) * 3;
            const float lx = (xp[0] + 0.5f) * 15.0f;
            const float ly = (xp[1] + 0.5f) * 15.0f;
            const float lz = (xp[2] + 0.5f) * 15.0f;
            const float fx = floorf(lx), fy = floorf(ly), fz = floorf(lz);
            const float tx = lx - fx,   ty = ly - fy,   tz = lz - fz;
            const int ix0 = (int)fx, iy0 = (int)fy, iz0 = (int)fz;

            float a0 = 0.f, a1 = 0.f, a2 = 0.f, a3 = 0.f;
#pragma unroll
            for (int c = 0; c < 8; ++c) {
                const int dx = c & 1, dy = (c >> 1) & 1, dz = c >> 2;
                const int ix = ix0 + dx, iy = iy0 + dy, iz = iz0 + dz;
                float w = (dx ? tx : 1.f - tx) * (dy ? ty : 1.f - ty)
                        * (dz ? tz : 1.f - tz);
                const bool ok = (ix >= 0) & (ix < RESO) & (iy >= 0) &
                                (iy < RESO) & (iz >= 0) & (iz < RESO);
                w = ok ? w : 0.f;
                const int cx = min(max(ix, 0), RESO - 1);
                const int cy = min(max(iy, 0), RESO - 1);
                const int cz = min(max(iz, 0), RESO - 1);
                const float4 v = *(const float4*)
                    (g + (((cz * RESO + cy) * RESO + cx) * FEAT + hi * 4));
                a0 += w * v.x; a1 += w * v.y; a2 += w * v.z; a3 += w * v.w;
            }
            v4h fv;
            fv[0] = (half_t)a0; fv[1] = (half_t)a1;
            fv[2] = (half_t)a2; fv[3] = (half_t)a3;
            *(v4h*)&sF[wave][ln][hi * 4] = fv;     // one ds_store_b64
        }
        __syncthreads();

        // ---------- layer 0: K=8 (zero-padded to 32), 4 N-tiles ----------
        {
            const half_t* ap = &sF[wave][ln][hi * 16];
            const v16h a = CAT16(*(const v8h*)ap, *(const v8h*)(ap + 8));
#pragma unroll
            for (int n = 0; n < 4; ++n) {
                const half_t* bp = f0b + n * FROW;
                const v16h bm = CAT16(*(const v8h*)bp, *(const v8h*)(bp + 8));
                v8f acc;
                const float bv = sB0[n * 16 + ln];
#pragma unroll
                for (int r = 0; r < 8; ++r) acc[r] = bv;
                acc = __builtin_amdgcn_wmma_f32_16x16x32_f16(
                    false, a, false, bm, (short)0, acc, false, false);
#pragma unroll
                for (int r = 0; r < 8; ++r) {
                    float v = acc[r];
                    v = fmaxf(v, 0.01f * v);
                    sH0[wave][hi * 8 + r][n * 16 + ln] = (half_t)v;
                }
            }
        }
        __syncthreads();

        // ---------- layer 1: K=64 (2 K-tiles), 4 N-tiles ----------
        {
            v8f acc[4];
#pragma unroll
            for (int n = 0; n < 4; ++n) {
                const float bv = sB1[n * 16 + ln];
#pragma unroll
                for (int r = 0; r < 8; ++r) acc[n][r] = bv;
            }
#pragma unroll
            for (int kt = 0; kt < 2; ++kt) {
                const half_t* ap = &sH0[wave][ln][kt * 32 + hi * 16];
                const v16h a = CAT16(*(const v8h*)ap, *(const v8h*)(ap + 8));
#pragma unroll
                for (int n = 0; n < 4; ++n) {
                    const half_t* bp = f1b + (kt * 8 + n) * FROW;
                    const v16h bm = CAT16(*(const v8h*)bp, *(const v8h*)(bp + 8));
                    acc[n] = __builtin_amdgcn_wmma_f32_16x16x32_f16(
                        false, a, false, bm, (short)0, acc[n], false, false);
                }
            }
#pragma unroll
            for (int n = 0; n < 4; ++n)
#pragma unroll
                for (int r = 0; r < 8; ++r) {
                    float v = acc[n][r];
                    v = fmaxf(v, 0.01f * v);
                    sH1[wave][hi * 8 + r][n * 16 + ln] = (half_t)v;
                }
        }
        __syncthreads();

        // ---------- layer 2: K=64 (2 K-tiles), N=16; fp32 store ----------
        {
            v8f acc;
            const float bv = sBo[ln];
#pragma unroll
            for (int r = 0; r < 8; ++r) acc[r] = bv;
#pragma unroll
            for (int kt = 0; kt < 2; ++kt) {
                const half_t* ap = &sH1[wave][ln][kt * 32 + hi * 16];
                const v16h a = CAT16(*(const v8h*)ap, *(const v8h*)(ap + 8));
                const half_t* bp = f2b + kt * 2 * FROW;
                const v16h bm = CAT16(*(const v8h*)bp, *(const v8h*)(bp + 8));
                acc = __builtin_amdgcn_wmma_f32_16x16x32_f16(
                    false, a, false, bm, (short)0, acc, false, false);
            }
#pragma unroll
            for (int r = 0; r < 8; ++r) {
                const size_t p = (size_t)bex * NPTS + p0 + hi * 8 + r;
                out[p * OUTD + ln] = acc[r];
            }
        }
        __syncthreads();   // protect sF/sH0/sH1 before next tile overwrites
    }
}

extern "C" void kernel_launch(void* const* d_in, const int* in_sizes, int n_in,
                              void* d_out, int out_size, void* d_ws, size_t ws_size,
                              hipStream_t stream) {
    const int*   idx = (const int*)  d_in[0];
    const float* x   = (const float*)d_in[1];
    const float* emb = (const float*)d_in[2];
    const float* W0  = (const float*)d_in[3];
    const float* b0  = (const float*)d_in[4];
    const float* W1  = (const float*)d_in[5];
    const float* b1  = (const float*)d_in[6];
    const float* Wo  = (const float*)d_in[7];
    const float* bo  = (const float*)d_in[8];
    float* out = (float*)d_out;

    const int blocks = NEXMP * BPE;   // 64 * 16 = 1024
    featmlp_kernel<<<dim3(blocks), dim3(TPB), 0, stream>>>(
        idx, x, emb, W0, b0, W1, b1, Wo, bo, out);
}
